// DimBlock_1_90563680403819
// MI455X (gfx1250) — compile-verified, matrix operand order checked by reading, not run on hardware
//
#include <hip/hip_runtime.h>
#include <hip/hip_bf16.h>
#include <stdint.h>

// ---------------------------------------------------------------------------
// Implicit-GEMM conv for MI455X (gfx1250):
//   O[b,oc,y,x] = sum_{c,kh,kw} X[b,c,y+kh,x+kw] * W[oc,c,kh,kw] + bias[oc]
// M=100 oc (7x16 tiles), K=2025 (64x32), N-tile = 64 pixels per workgroup.
// fp32 via bf16 hi/lo split: x*w ~= xh*wh + xh*wl + xl*wh
// -> 3x v_wmma_f32_16x16x32_bf16 per (K-chunk, pixel set), f32 accumulate.
// Input patch staged to LDS by one Tensor Data Mover descriptor
// (tensor_load_to_lds + s_wait_tensorcnt, confirmed in disasm); B-tile
// im2col uses register-resident incremental addressing + double-buffered LDS.
// ---------------------------------------------------------------------------

#if __has_include(<hip/amd_detail/amd_gfx1250_TDM.h>)
#define TDM_HDR 1
#else
#define TDM_HDR 0
#endif
#if __has_builtin(__builtin_amdgcn_tensor_load_to_lds)
#define TDM_BUILTIN 1
#else
#define TDM_BUILTIN 0
#endif

#if TDM_BUILTIN && !TDM_HDR
#define TDM_MODE 5          // ROCm lane: 5-arg builtin
#elif TDM_BUILTIN && TDM_HDR
#define TDM_MODE 6          // therock lane: 6-arg builtin (confirmed round 2/3)
#else
#define TDM_MODE 0          // fallback: plain vector copies
#endif

typedef __attribute__((ext_vector_type(16))) __bf16 v16bf;
typedef __attribute__((ext_vector_type(8)))  float  v8f;
typedef __attribute__((ext_vector_type(4)))  unsigned int v4u;
typedef __attribute__((ext_vector_type(8)))  int    v8i;
typedef __attribute__((ext_vector_type(4)))  int    v4i;

#define N_B   8
#define N_C   25
#define HW    256
#define OC    100
#define OCT   7        // ceil(100/16)
#define KTOT  2025
#define NKC   64       // ceil(2025/32)
#define OHW   248
#define NPIX  64       // pixels per workgroup
#define NSET  4        // NPIX/16 B-fragment sets
#define XT    4        // x-tiles per row (4*64=256 >= 248)
#define LDSW  72       // input patch row width (64 + 8 halo)
#define TPB   224      // 7 waves
#define NSTG  ((NSET * 512 + TPB - 1) / TPB)   // 10 staging slots per thread

// K-element mapping inside a 16-bit WMMA fragment (A rows and B cols):
// lane<16 holds K {0..7,16..23}, lane>=16 holds K {8..15,24..31}.
__device__ __forceinline__ int kmap(int lane, int e) {
    return (lane < 16) ? (e < 8 ? e : e + 8) : (e < 8 ? e + 8 : e + 16);
}

// ---------------------------------------------------------------------------
// Pre-pass: reorder weights into WMMA A-fragment layout, bf16 hi/lo planes.
// ws layout: frag[(tile*NKC + kc)*2 + sel][lane*16 + e], sel 0=hi, 1=lo.
// ---------------------------------------------------------------------------
__global__ void prep_weights(const float* __restrict__ w, __bf16* __restrict__ wsA) {
    int idx = blockIdx.x * 256 + threadIdx.x;           // over 7*64*32*16 items
    if (idx >= OCT * NKC * 32 * 16) return;
    int e    = idx & 15;
    int lane = (idx >> 4) & 31;
    int kc   = (idx >> 9) & 63;
    int tile = idx >> 15;
    int oc   = tile * 16 + (lane & 15);
    int k    = kc * 32 + kmap(lane, e);
    float x  = (oc < OC && k < KTOT) ? w[(size_t)oc * KTOT + k] : 0.0f;
    __bf16 hi = (__bf16)x;
    __bf16 lo = (__bf16)(x - (float)hi);
    size_t base = ((size_t)(tile * NKC + kc) * 2) * 512 + (size_t)lane * 16 + e;
    wsA[base]       = hi;
    wsA[base + 512] = lo;
}

// ---------------------------------------------------------------------------
// Main kernel: one workgroup = 64-pixel tile (b, y, x0), 7 waves x 16 oc.
// ---------------------------------------------------------------------------
__launch_bounds__(TPB)
__global__ void conv_wmma(const float* __restrict__ x,
                          const __bf16* __restrict__ wsA,
                          const float* __restrict__ bias,
                          float* __restrict__ out) {
    __shared__ float ldsX[N_C * 9 * LDSW];                       // 64800 B
    __shared__ __attribute__((aligned(32))) __bf16 bufB[2][2][NSET * 512]; // 16 KB

    const int tid  = threadIdx.x;
    const int bid  = blockIdx.x;
    const int xt   = bid & (XT - 1);
    const int y    = (bid >> 2) % OHW;
    const int b    = bid / (XT * OHW);
    const int x0   = xt * NPIX;
    const int lane = tid & 31;
    const int wv   = tid >> 5;                             // 0..6 oc tile

    const float* rowbase = x + (size_t)b * (N_C * HW * HW) + (size_t)y * HW;

#if TDM_MODE
    if (tid < 32) {                       // wave 0 issues one TDM descriptor
        uint64_t ga = (uint64_t)(uintptr_t)(rowbase + x0);
        unsigned int lds0 = (unsigned int)(uintptr_t)(&ldsX[0]);
        int td0 = HW - x0;                // clip -> OOB columns read as zero
        int td1 = HW - y;
        v4u g0; g0[0] = 1u;  g0[1] = lds0;
        g0[2] = (unsigned int)ga;
        g0[3] = (unsigned int)((ga >> 32) & 0x1FFFFFFu) | 0x80000000u; // type=2
        v8i g1;
        g1[0] = 0x00020000;                               // data_size = 4B
        g1[1] = (td0 & 0xFFFF) << 16;                     // tensor_dim0 lo16
        g1[2] = ((td0 >> 16) & 0xFFFF) | ((td1 & 0xFFFF) << 16);
        g1[3] = ((td1 >> 16) & 0xFFFF) | (LDSW << 16);    // tile_dim0 = 72
        g1[4] = 9 | (N_C << 16);                          // tile_dim1,2 = 9,25
        g1[5] = HW;                                       // dim0 stride = 256
        g1[6] = 0;
        g1[7] = (HW * HW) >> 16;                          // dim1 stride = 65536
        v4i g2; g2[0] = N_C; g2[1] = 0; g2[2] = HW * HW; g2[3] = 0;
        v4i g3; g3[0] = 0;   g3[1] = 0; g3[2] = 0;       g3[3] = 0;
#if TDM_MODE == 5
        __builtin_amdgcn_tensor_load_to_lds(g0, g1, g2, g3, 0);
#else
        v8i g4 = {0, 0, 0, 0, 0, 0, 0, 0};                // unused VADDR4 group
        __builtin_amdgcn_tensor_load_to_lds(g0, g1, g2, g3, g4, 0);
#endif
#if __has_builtin(__builtin_amdgcn_s_wait_tensorcnt)
        __builtin_amdgcn_s_wait_tensorcnt(0);
#else
        asm volatile("s_wait_tensorcnt 0" ::: "memory");
#endif
    }
#else
    // Fallback: cooperative b128 copies (rows c*9+kh, 72 f32 columns each)
    for (int it = tid; it < N_C * 9 * (LDSW / 4); it += TPB) {
        int r = it / (LDSW / 4), j = it - r * (LDSW / 4);
        int c = r / 9, kh = r - c * 9;
        int cs = x0 + 4 * j; if (cs > HW - 4) cs = HW - 4;
        const float4* gp = (const float4*)(rowbase + (size_t)c * (HW * HW) +
                                           (size_t)kh * HW + cs);
        *(float4*)&ldsX[r * LDSW + 4 * j] = *gp;
    }
#endif

    // --- register-resident incremental im2col addressing -------------------
    // k = r*9 + kw (r = c*9+kh); per K-chunk k += 32 => r += 3, kw += 5 (wrap).
    int kkv[NSTG], kwv[NSTG], offv[NSTG];
#pragma unroll
    for (int j = 0; j < NSTG; ++j) {
        int it = tid + j * TPB;
        if (it < NSET * 512) {
            int e = it & 15, bl = (it >> 4) & 31, s = it >> 9;
            int n = s * 16 + (bl & 15);
            int kk = kmap(bl, e);
            int r = kk / 9, kw = kk - r * 9;
            kkv[j] = kk;
            kwv[j] = kw;
            offv[j] = r * LDSW + kw + n;
        }
    }

    auto stage = [&](int kc, int pb) {
        int kcb = kc * 32;
#pragma unroll
        for (int j = 0; j < NSTG; ++j) {
            int it = tid + j * TPB;
            if (it < NSET * 512) {
                int k = kcb + kkv[j];
                float v = 0.0f;
                if (k < KTOT) v = ldsX[offv[j]];
                __bf16 hi = (__bf16)v;
                bufB[pb][0][it] = hi;
                bufB[pb][1][it] = (__bf16)(v - (float)hi);
                int kw = kwv[j] + 5, off = offv[j] + 3 * LDSW + 5;
                if (kw >= 9) { kw -= 9; off += LDSW - 9; }
                kwv[j] = kw; offv[j] = off;
            }
        }
    };

    v8f acc[NSET];
#pragma unroll
    for (int s = 0; s < NSET; ++s)
        acc[s] = (v8f){0.f, 0.f, 0.f, 0.f, 0.f, 0.f, 0.f, 0.f};

    __syncthreads();          // ldsX (TDM or copies) visible to all 7 waves
    stage(0, 0);              // stage chunk 0 exactly once (address state!)
    __syncthreads();

    for (int kc = 0; kc < NKC; ++kc) {
        int p = kc & 1;
        if (kc + 1 < NKC) stage(kc + 1, p ^ 1);           // overlaps WMMA

        const __bf16* fa = wsA + (size_t)((wv * NKC + kc) * 2) * 512 + lane * 16;
        __builtin_prefetch(fa + 1024, 0, 1);              // next chunk -> cache
        v16bf a_hi = *(const v16bf*)fa;
        v16bf a_lo = *(const v16bf*)(fa + 512);
#pragma unroll
        for (int s = 0; s < NSET; ++s) {
            v16bf b_hi = *(const v16bf*)&bufB[p][0][s * 512 + lane * 16];
            v16bf b_lo = *(const v16bf*)&bufB[p][1][s * 512 + lane * 16];
            acc[s] = __builtin_amdgcn_wmma_f32_16x16x32_bf16(
                false, a_hi, false, b_hi, (short)0, acc[s], false, false);
            acc[s] = __builtin_amdgcn_wmma_f32_16x16x32_bf16(
                false, a_hi, false, b_lo, (short)0, acc[s], false, false);
            acc[s] = __builtin_amdgcn_wmma_f32_16x16x32_bf16(
                false, a_lo, false, b_hi, (short)0, acc[s], false, false);
        }
        __syncthreads();
    }

    // --- store: D layout lane/VGPR -> (M=oc row, N=pixel) ------------------
    const int nn = lane & 15;
#pragma unroll
    for (int s = 0; s < NSET; ++s) {
        int xc = x0 + s * 16 + nn;
        if (xc < OHW) {
#pragma unroll
            for (int v = 0; v < 8; ++v) {
                int m  = (lane < 16) ? v : v + 8;
                int oc = wv * 16 + m;
                if (oc < OC) {
                    size_t o = (((size_t)b * OC + oc) * OHW + y) * OHW + xc;
                    out[o] = acc[s][v] + bias[oc];
                }
            }
        }
    }
}

extern "C" void kernel_launch(void* const* d_in, const int* in_sizes, int n_in,
                              void* d_out, int out_size, void* d_ws, size_t ws_size,
                              hipStream_t stream) {
    const float* x    = (const float*)d_in[0];
    const float* w    = (const float*)d_in[1];
    const float* bias = (const float*)d_in[2];
    float*       out  = (float*)d_out;
    __bf16*      wsA  = (__bf16*)d_ws;   // needs 7*64*2*512*2 B = 917504 B

    int prep_items = OCT * NKC * 32 * 16;
    prep_weights<<<(prep_items + 255) / 256, 256, 0, stream>>>(w, wsA);

    dim3 grid(N_B * OHW * XT);           // 8*248*4 = 7936 workgroups
    conv_wmma<<<grid, TPB, 0, stream>>>(x, wsA, bias, out);
}
